// NewCompoundMoE_1984274891215
// MI455X (gfx1250) — matrix operand order repaired and checked by reading, not compiled, not run on hardware
//
#include <hip/hip_runtime.h>
#include <hip/hip_bf16.h>

// ---------------- types ----------------
typedef __attribute__((ext_vector_type(16))) __bf16 v16bf;
typedef __attribute__((ext_vector_type(8)))  __bf16 v8bf;
typedef __attribute__((ext_vector_type(8)))  float  v8f;
typedef __attribute__((ext_vector_type(4)))  unsigned int u32x4;
typedef __attribute__((ext_vector_type(8)))  unsigned int u32x8;

#define T_TOK   1024
#define DIM     1024
#define FFN     512
#define NEXP    16
#define NOUT    8

// TDM pads 4 DWORDs every 256 DWORDs -> x-half tile row pitch = 512+8 = 520 DW.
// offset(elem n) = n + (n>>8)*4  (rows are exactly 2 pad intervals long)
#define XPITCH_DW  520
// h tile (bf16): row pitch 520 halves = 1040B = 260 DW, 260 % 64banks = 4 -> conflict-free
#define LDSPH      520

// ---------------- fragment loaders ----------------
// A-fragment (16xK chunk of 32) from the TDM-padded f32 LDS tile.
// Lane l: M = l&15, hi = l>>4; elems 0..7 = K[kk+hi*16..), elems 8..15 = K[kk+hi*16+8..)
static __device__ inline v16bf load_a_frag_f32(const float* buf, int row, int kk, int hi) {
  int c0 = kk + hi * 16;
  int n0 = row * 512 + c0;
  int o0 = n0 + ((n0 >> 8) << 2);          // TDM pad: +4 DW per 256 DW
  int c1 = c0 + 8;
  int n1 = row * 512 + c1;
  int o1 = n1 + ((n1 >> 8) << 2);
  v8f f0 = *(const v8f*)(buf + o0);
  v8f f1 = *(const v8f*)(buf + o1);
  v16bf a;
#pragma unroll
  for (int i = 0; i < 8; ++i) { a[i] = (__bf16)f0[i]; a[i + 8] = (__bf16)f1[i]; }
  return a;
}

// A-fragment from a row-major bf16 LDS row (h tile, pitch LDSPH halves)
static __device__ inline v16bf load_a_frag_bf16(const __bf16* row, int k0, int hi) {
  v8bf a0 = *(const v8bf*)(row + k0 + hi * 16);
  v8bf a1 = *(const v8bf*)(row + k0 + hi * 16 + 8);
  v16bf a;
#pragma unroll
  for (int i = 0; i < 8; ++i) { a[i] = a0[i]; a[i + 8] = a1[i]; }
  return a;
}

// B-fragment (Kx16 chunk of 32) from a row-major f32 weight row W[n][*]:
// lane l: N = l&15, hi = l>>4; elem i = W_row[koff + i] with koff = k0 + hi*16.
static __device__ inline v16bf load_b_frag(const float* wrow, int koff) {
  const float4* p = (const float4*)(wrow + koff);
  v16bf b;
#pragma unroll
  for (int q = 0; q < 4; ++q) {
    float4 v = p[q];
    b[q * 4 + 0] = (__bf16)v.x;
    b[q * 4 + 1] = (__bf16)v.y;
    b[q * 4 + 2] = (__bf16)v.z;
    b[q * 4 + 3] = (__bf16)v.w;
  }
  return b;
}

// ---------------- kernel 0: zero the output ----------------
__global__ void zero_out_kernel(float* __restrict__ out, int n) {
  int i = blockIdx.x * blockDim.x + threadIdx.x;
  if (i < n) out[i] = 0.f;
}

// ---------------- kernel 1: router ----------------
__global__ __launch_bounds__(256)
void router_kernel(const float* __restrict__ x, const float* __restrict__ wog,
                   const float* __restrict__ wig, float* __restrict__ cw,
                   float* __restrict__ selw_out) {
  const int wv   = threadIdx.x >> 5;
  const int lane = threadIdx.x & 31;
  const int t    = blockIdx.x * 8 + wv;   // one wave per token

  float acc[24];
#pragma unroll
  for (int r = 0; r < 24; ++r) acc[r] = 0.f;

  const float* xr = x + (size_t)t * DIM;
  for (int d = lane; d < DIM; d += 32) {
    float xv = xr[d];
#pragma unroll
    for (int r = 0; r < NOUT; ++r) acc[r] += xv * wog[r * DIM + d];
#pragma unroll
    for (int r = 0; r < 16; ++r) acc[8 + r] += xv * wig[r * DIM + d];
  }
#pragma unroll
  for (int off = 16; off > 0; off >>= 1) {
#pragma unroll
    for (int r = 0; r < 24; ++r) acc[r] += __shfl_xor(acc[r], off, 32);
  }

  if (lane == 0) {
    float mx = acc[0];
#pragma unroll
    for (int r = 1; r < NOUT; ++r) mx = fmaxf(mx, acc[r]);
    float p[NOUT];
#pragma unroll
    for (int r = 0; r < NOUT; ++r) p[r] = __expf(acc[r] - mx);
    int s0 = 0;
#pragma unroll
    for (int r = 1; r < NOUT; ++r) if (p[r] > p[s0]) s0 = r;
    int s1 = (s0 == 0) ? 1 : 0;
#pragma unroll
    for (int r = 0; r < NOUT; ++r) if (r != s0 && p[r] > p[s1]) s1 = r;
    float w0 = p[s0], w1 = p[s1];
    float sw = w0 + w1;
    w0 /= sw; w1 /= sw;
    float i00 = acc[8 + s0 * 2 + 0], i01 = acc[8 + s0 * 2 + 1];
    int a0 = (i01 > i00) ? 1 : 0;
    int e0 = s0 * 2 + a0;
    int e1 = s0 * 2 + (1 - a0);
    float i10 = acc[8 + s1 * 2 + 0], i11 = acc[8 + s1 * 2 + 1];
    int e2 = s1 * 2 + ((i11 > i10) ? 1 : 0);

    float* cwt = cw + t * NEXP;
#pragma unroll
    for (int e = 0; e < NEXP; ++e) cwt[e] = 0.f;
    cwt[e0] += w0; cwt[e1] += w0; cwt[e2] += w1;

    float* s3 = selw_out + t * 3;
    s3[0] = w0; s3[1] = w0; s3[2] = w1;
  }
}

// ---------------- kernel 2: deterministic per-expert token lists ----------------
__global__ void build_lists_kernel(const float* __restrict__ cw, int* __restrict__ cnt,
                                   int* __restrict__ lists) {
  const int e    = blockIdx.x;   // one wave32 per expert
  const int lane = threadIdx.x;
  int count = 0;
  int* le = lists + e * T_TOK;
  for (int base = 0; base < T_TOK; base += 32) {
    int t = base + lane;
    bool sel = (cw[t * NEXP + e] != 0.f);
    unsigned long long m = __ballot(sel);
    int prefix = __popcll(m & ((1ull << lane) - 1ull));
    if (sel) le[count + prefix] = t;
    count += __popcll(m);
  }
  if (lane == 0) cnt[e] = count;
}

// ---------------- kernel 3: fused expert FFN with TDM token gather --------------
__global__ __launch_bounds__(256)
void expert_kernel(const float* __restrict__ x, const float* __restrict__ wg,
                   const float* __restrict__ wu, const float* __restrict__ wd,
                   const float* __restrict__ cw, const int* __restrict__ cnt,
                   const int* __restrict__ lists, float* __restrict__ out) {
  // One buffer, two lives: f32 x-half tile (TDM dest, padded pitch 520 DW),
  // then bf16 h tile (pitch 520 halves) for the down projection.
  __shared__ float xtile[16 * XPITCH_DW];
  __shared__ int   toks[16];
  __shared__ float wgts[16];

  const int e     = blockIdx.x;
  const int tile  = blockIdx.y;
  const int count = cnt[e];
  if (tile * 16 >= count) return;      // block-uniform: EXEC all-1s below

  const int wv   = threadIdx.x >> 5;   // 8 waves
  const int lane = threadIdx.x & 31;
  const int nl   = lane & 15;
  const int hi   = lane >> 4;

  if (threadIdx.x < 16) {
    int idx = tile * 16 + threadIdx.x;
    int t = (idx < count) ? lists[e * T_TOK + idx] : -1;
    toks[threadIdx.x] = t;
    wgts[threadIdx.x] = (t >= 0) ? cw[t * NEXP + e] : 0.f;
  }

  // ---- build uniform (SGPR) gather indices for the TDM descriptor ----
  // All addressing below is block-uniform; readfirstlane pins the packed words
  // to SGPRs so the inline tensor_load_to_lds gets s-register descriptor groups.
  unsigned idxw[8];
#pragma unroll
  for (int q = 0; q < 8; ++q) {
    int i0 = tile * 16 + q * 2;
    int i1 = i0 + 1;
    int t0 = (i0 < count) ? lists[e * T_TOK + i0] : 0;
    int t1 = (i1 < count) ? lists[e * T_TOK + i1] : 0;
    idxw[q] = (unsigned)__builtin_amdgcn_readfirstlane(
        (int)(((unsigned)t1 << 16) | (unsigned)t0));
  }
  __syncthreads();

  // ---- phase 1: g = x*Wg^T, u = x*Wu^T (each wave owns 64 of F=512 cols),
  //      K=D=1024 streamed in two 512-col halves gathered by the TDM ----
  v8f gacc[4], uacc[4];
#pragma unroll
  for (int tn = 0; tn < 4; ++tn) { gacc[tn] = {}; uacc[tn] = {}; }
  const int nbase = wv * 64;

  for (int kh = 0; kh < 2; ++kh) {
    if (wv == 0) {
      // ---- Tensor DMA descriptor (D#): gather 16 token rows x 512 f32 cols ----
      unsigned long long ga =
          (unsigned long long)(uintptr_t)x + (unsigned long long)kh * (512ull * 4ull);
      unsigned int lds_base = (unsigned int)(uintptr_t)&xtile[0];
      u32x4 g0;
      g0[0] = 0x80000001u;                 // gather_mode=1, 16-bit indices, count=1
      g0[1] = lds_base;                    // lds_addr
      g0[2] = (unsigned int)(ga & 0xFFFFFFFFull);              // global_addr lo
      g0[3] = ((unsigned int)(ga >> 32) & 0x01FFFFFFu) | 0x80000000u; // hi | type=2
      u32x8 g1;
      g1[0] = 0x07D20000u;   // data_size=4B | pad_enable | pad_interval=256DW | pad=4DW
      g1[1] = (512u  << 16); // tensor_dim0 = 512 (this K-half)
      g1[2] = (1024u << 16); // tensor_dim0 hi=0 | tensor_dim1 = 1024 rows
      g1[3] = (512u  << 16); // tensor_dim1 hi=0 | tile_dim0 = 512
      g1[4] = 16u;           // tile_dim1 = #row indices
      g1[5] = 1024u;         // tensor_dim0_stride = D elements (lo 32)
      g1[6] = 0u;
      g1[7] = 0u;
      u32x4 g2, g3;
#pragma unroll
      for (int q = 0; q < 4; ++q) {
        g2[q] = idxw[q];       // row_index pairs 0..7
        g3[q] = idxw[4 + q];   // row_index pairs 8..15
      }
      asm volatile("tensor_load_to_lds %0, %1, %2, %3"
                   :: "s"(g0), "s"(g1), "s"(g2), "s"(g3)
                   : "memory");
      __builtin_amdgcn_s_wait_tensorcnt(0);
    }
    __syncthreads();   // DMA done; tile visible to all 8 waves

    for (int kk = 0; kk < 512; kk += 32) {
      v16bf a = load_a_frag_f32(xtile, nl, kk, hi);
      const int k0 = kh * 512 + kk;
#pragma unroll
      for (int tn = 0; tn < 4; ++tn) {
        const float* gr = wg + ((size_t)(e * FFN + nbase + tn * 16 + nl)) * DIM;
        const float* ur = wu + ((size_t)(e * FFN + nbase + tn * 16 + nl)) * DIM;
        __builtin_prefetch(gr + k0 + 32, 0, 3);   // global_prefetch_b8
        __builtin_prefetch(ur + k0 + 32, 0, 3);
        v16bf bg = load_b_frag(gr, k0 + hi * 16);
        gacc[tn] = __builtin_amdgcn_wmma_f32_16x16x32_bf16(
            false, a, false, bg, (short)0, gacc[tn], false, false);
        v16bf bu = load_b_frag(ur, k0 + hi * 16);
        uacc[tn] = __builtin_amdgcn_wmma_f32_16x16x32_bf16(
            false, a, false, bu, (short)0, uacc[tn], false, false);
      }
    }
    __syncthreads();   // all waves done reading this half before next DMA / reuse
  }

  // ---- SwiGLU in f32 accumulators; write bf16 h tile into the reused buffer ----
  __bf16* lh = (__bf16*)xtile;
#pragma unroll
  for (int tn = 0; tn < 4; ++tn) {
    int n = nbase + tn * 16 + nl;
#pragma unroll
    for (int r = 0; r < 8; ++r) {
      int m = r + hi * 8;
      float g = gacc[tn][r];
      float u = uacc[tn][r];
      float h = (g / (1.f + __expf(-g))) * u;   // silu(g)*u
      lh[m * LDSPH + n] = (__bf16)h;
    }
  }
  __syncthreads();

  // ---- phase 2: y = h * Wd^T (each wave owns 128 of D=1024 cols), K=F=512 ----
  v8f yacc[8];
#pragma unroll
  for (int tn = 0; tn < 8; ++tn) yacc[tn] = {};

  const int dbase = wv * 128;
  const __bf16* hrow = lh + nl * LDSPH;
  for (int k0 = 0; k0 < FFN; k0 += 32) {
    v16bf a = load_a_frag_bf16(hrow, k0, hi);
#pragma unroll
    for (int tn = 0; tn < 8; ++tn) {
      const float* dr = wd + ((size_t)(e * DIM + dbase + tn * 16 + nl)) * FFN;
      __builtin_prefetch(dr + k0 + 32, 0, 3);
      v16bf bd = load_b_frag(dr, k0 + hi * 16);
      yacc[tn] = __builtin_amdgcn_wmma_f32_16x16x32_bf16(
          false, a, false, bd, (short)0, yacc[tn], false, false);
    }
  }

  // ---- weighted combine into output (f32 atomics across experts) ----
#pragma unroll
  for (int tn = 0; tn < 8; ++tn) {
    int dcol = dbase + tn * 16 + nl;
#pragma unroll
    for (int r = 0; r < 8; ++r) {
      int m = r + hi * 8;
      int t = toks[m];
      if (t >= 0) atomicAdd(out + (size_t)t * DIM + dcol, wgts[m] * yacc[tn][r]);
    }
  }
}

// ---------------- launch ----------------
extern "C" void kernel_launch(void* const* d_in, const int* in_sizes, int n_in,
                              void* d_out, int out_size, void* d_ws, size_t ws_size,
                              hipStream_t stream) {
  (void)in_sizes; (void)n_in; (void)out_size; (void)ws_size;
  const float* x   = (const float*)d_in[0];   // [1,1024,1024]
  const float* wog = (const float*)d_in[1];   // [8,1024]
  const float* wig = (const float*)d_in[2];   // [8,2,1024]
  const float* wgp = (const float*)d_in[3];   // [16,512,1024]
  const float* wup = (const float*)d_in[4];   // [16,512,1024]
  const float* wdp = (const float*)d_in[5];   // [16,1024,512]
  float* out = (float*)d_out;                 // [T*D] out ++ [T*3] sel_w

  // workspace: cw [T*16] f32 | cnt [16] int (256B slot) | lists [16*T] int
  float* cw  = (float*)d_ws;
  char*  p   = (char*)d_ws + (size_t)T_TOK * NEXP * sizeof(float);
  int*   cnt   = (int*)p;
  int*   lists = (int*)(p + 256);

  zero_out_kernel<<<(T_TOK * DIM) / 256, 256, 0, stream>>>(out, T_TOK * DIM);
  router_kernel<<<T_TOK / 8, 256, 0, stream>>>(x, wog, wig, cw, out + (size_t)T_TOK * DIM);
  build_lists_kernel<<<NEXP, 32, 0, stream>>>(cw, cnt, lists);
  dim3 grid(NEXP, T_TOK / 16);
  expert_kernel<<<grid, 256, 0, stream>>>(x, wgp, wup, wdp, cw, cnt, lists, out);
}
